// FlexMoERouter_82274393522254
// MI455X (gfx1250) — matrix operand order repaired and checked by reading, not compiled, not run on hardware
//
#include <hip/hip_runtime.h>
#include <hip/hip_bf16.h>
#include <stdint.h>

// ---------------------------------------------------------------------------
// FlexMoE router for MI455X (gfx1250), wave32 + bf16 WMMA + async->LDS staging.
// M=16384 rows, H=2048, E=64 experts, TOP_K=4.
// HBM floor: 537MB activations @ 23.3TB/s => ~23us. bf16 WMMA keeps matrix
// math far below that. Weights are staged into LDS once per workgroup with
// GLOBAL_LOAD_ASYNC_TO_LDS (double-buffered, ASYNCcnt-pipelined) so all 4
// waves share one L2 read stream; activation loads are non-temporal so the
// one-pass 537MB stream does not evict L2-resident weights.
// ---------------------------------------------------------------------------

#define Hdim 2048
#define Edim 64
#define Mrows 16384
#define KTILES 64          // Hdim / 32 (bf16 WMMA K)
#define LN_EPS 1e-5f

typedef __attribute__((ext_vector_type(16))) __bf16 v16bf;
typedef __attribute__((ext_vector_type(8)))  float  v8f;

// ---------------------------------------------------------------------------
// Workspace layout (dwords):
//   [0 .. 262143]   packed bf16 B-fragments: [m:4][kt:64][nt:4][lane:32][v:8]
//   [262144..]      cg[64], cb[64], btot[64], esum[64]
// ---------------------------------------------------------------------------
#define WS_WP    0
#define WS_CG    262144
#define WS_CB    262208
#define WS_BTOT  262272
#define WS_ESUM  262336

// ---------- pre-pass: pack weights into per-lane WMMA B-fragment order ------
// B (32x16 bf16) layout (mirrors documented sparse-B striping):
//   lanes 0-15  : column n = lane,    K = 0..7, 16..23  (2 packed per VGPR v=0..7)
//   lanes 16-31 : column n = lane-16, K = 8..15, 24..31
__global__ void pack_weights_kernel(const float* __restrict__ Wg,
                                    const float* __restrict__ Wi,
                                    const float* __restrict__ Wt,
                                    const float* __restrict__ Wa,
                                    const float* __restrict__ ln_g,
                                    uint32_t* __restrict__ Wp) {
  int t = blockIdx.x * blockDim.x + threadIdx.x;       // 0 .. 262143
  int v    = t & 7;
  int lane = (t >> 3) & 31;
  int nt   = (t >> 8) & 3;
  int kt   = (t >> 10) & 63;
  int m    = (t >> 16) & 3;
  int n = nt * 16 + (lane & 15);
  int k = kt * 32 + ((lane >> 4) << 3) + 2 * v + ((v >> 2) << 3);
  const float* W = (m == 0) ? Wg : (m == 1) ? Wi : (m == 2) ? Wt : Wa;
  float f0 = W[(size_t)k * Edim + n];
  float f1 = W[(size_t)(k + 1) * Edim + n];
  if (m == 0) { f0 *= ln_g[k]; f1 *= ln_g[k + 1]; }    // fold LN gamma into Wg
  __bf16 h0 = (__bf16)f0, h1 = (__bf16)f1;
  uint32_t lo = (uint32_t)__builtin_bit_cast(unsigned short, h0);
  uint32_t hi = (uint32_t)__builtin_bit_cast(unsigned short, h1);
  Wp[t] = lo | (hi << 16);
}

// ---------- pre-pass: LN correction constants + summed biases ---------------
__global__ void router_consts_kernel(const float* __restrict__ Wg,
                                     const float* __restrict__ ln_g,
                                     const float* __restrict__ ln_b,
                                     const float* __restrict__ bg,
                                     const float* __restrict__ bi,
                                     const float* __restrict__ bt,
                                     const float* __restrict__ ba,
                                     float* __restrict__ cg,
                                     float* __restrict__ cb,
                                     float* __restrict__ btot,
                                     float* __restrict__ esum) {
  int e = threadIdx.x;   // 64 threads
  float sg = 0.f, sb = 0.f;
  for (int h = 0; h < Hdim; ++h) {
    float w = Wg[(size_t)h * Edim + e];
    sg += ln_g[h] * w;
    sb += ln_b[h] * w;
  }
  cg[e] = sg;
  cb[e] = sb;
  btot[e] = bg[e] + bi[e] + bt[e] + ba[e];
  esum[e] = 0.f;   // zero per-call so graph replay is deterministic
}

// ---------- A-fragment loader: f32 -> bf16, non-temporal (stream-once) ------
template <bool STATS>
__device__ __forceinline__ v16bf mk_frag(const float* __restrict__ p,
                                         float& s, float& ss) {
  v8f c0 = __builtin_nontemporal_load((const v8f*)(p));        // K base..+7
  v8f c1 = __builtin_nontemporal_load((const v8f*)(p + 16));   // K base+16..+23
  v16bf a;
#pragma unroll
  for (int i = 0; i < 8; ++i) {
    float f0 = c0[i], f1 = c1[i];
    if (STATS) { s += f0 + f1; ss += f0 * f0 + f1 * f1; }
    a[i]     = (__bf16)f0;
    a[i + 8] = (__bf16)f1;
  }
  return a;
}

// ---------- async copy of one 16KB weight K-slice (4 matrices) into LDS -----
// 128 threads x 8 x 16B = 16KB. Tracked on ASYNCcnt.
__device__ __forceinline__ void issue_wslice(const uint32_t* __restrict__ Wp,
                                             int kt, int buf,
                                             uint32_t ldsBase, int tid) {
  const char* gbase = (const char*)Wp + (size_t)kt * 4096;
#pragma unroll
  for (int m = 0; m < 4; ++m) {
#pragma unroll
    for (int i = 0; i < 2; ++i) {
      int chunk = tid + i * 128;                               // 0..255
      uint64_t g = (uint64_t)(uintptr_t)(gbase + (size_t)m * 262144
                                               + (size_t)chunk * 16);
      uint32_t l = ldsBase + (uint32_t)(buf * 16384 + m * 4096 + chunk * 16);
      asm volatile("global_load_async_to_lds_b128 %0, %1, off"
                   :: "v"(l), "v"(g) : "memory");
    }
  }
}

// ---------- main fused kernel: GEMM + LN-fold + softmax + top4 --------------
__global__ __launch_bounds__(128)
void router_main_kernel(const float* __restrict__ X,
                        const float* __restrict__ Mi,
                        const float* __restrict__ Mt,
                        const float* __restrict__ Ma,
                        const uint32_t* __restrict__ Wp,
                        const float* __restrict__ cg,
                        const float* __restrict__ cb,
                        const float* __restrict__ btot,
                        float* __restrict__ esum,
                        int* __restrict__ outI,
                        float* __restrict__ outP) {
  __shared__ __align__(16) unsigned char smem[32768];   // 2 x 16KB W slices

  const int tid    = threadIdx.x;
  const int lane   = tid & 31;
  const int wave   = tid >> 5;
  const int laneHi = lane >> 4;
  const int n0     = lane & 15;
  const int rowBase = blockIdx.x * 64 + wave * 16;
  const int myRow   = rowBase + n0;          // row whose A-slice this lane loads
  const int coff    = laneHi ? 8 : 0;

  const float* xr = X  + (size_t)myRow * Hdim;
  const float* ir = Mi + (size_t)myRow * Hdim;
  const float* tr = Mt + (size_t)myRow * Hdim;
  const float* ar = Ma + (size_t)myRow * Hdim;

  v8f accX[4], accM[4];
#pragma unroll
  for (int nt = 0; nt < 4; ++nt) { accX[nt] = (v8f)0.f; accM[nt] = (v8f)0.f; }

  float s = 0.f, ss = 0.f;   // raw-x row statistics (this lane's half-row)

  const uint32_t ldsBase = (uint32_t)(uintptr_t)smem;
  const uint32_t fragOff = (uint32_t)(lane * 32);      // this lane's B frag

  issue_wslice(Wp, 0, 0, ldsBase, tid);                // prologue: slice 0

  for (int kt = 0; kt < KTILES; ++kt) {
    if (kt + 1 < KTILES) {
      issue_wslice(Wp, kt + 1, (kt + 1) & 1, ldsBase, tid);
      asm volatile("s_wait_asynccnt 0x8" ::: "memory"); // slice kt landed
    } else {
      asm volatile("s_wait_asynccnt 0x0" ::: "memory");
    }
    __syncthreads();                                    // all waves' copies in

    const unsigned char* wbuf = smem + (kt & 1) * 16384;
    const int kb = kt * 32 + coff;

    __builtin_prefetch(xr + kb + 32, 0, 0);             // global_prefetch_b8

    // x @ (g .* Wg)  -- LN-corrected in epilogue
    {
      v16bf a = mk_frag<true>(xr + kb, s, ss);
#pragma unroll
      for (int nt = 0; nt < 4; ++nt) {
        v16bf b = *(const v16bf*)(wbuf + nt * 1024 + fragOff);
        accX[nt] = __builtin_amdgcn_wmma_f32_16x16x32_bf16(
            false, a, false, b, (short)0, accX[nt], false, false);
      }
    }
    // three modality routers on raw tensors
    {
      v16bf a = mk_frag<false>(ir + kb, s, ss);
#pragma unroll
      for (int nt = 0; nt < 4; ++nt) {
        v16bf b = *(const v16bf*)(wbuf + 4096 + nt * 1024 + fragOff);
        accM[nt] = __builtin_amdgcn_wmma_f32_16x16x32_bf16(
            false, a, false, b, (short)0, accM[nt], false, false);
      }
    }
    {
      v16bf a = mk_frag<false>(tr + kb, s, ss);
#pragma unroll
      for (int nt = 0; nt < 4; ++nt) {
        v16bf b = *(const v16bf*)(wbuf + 8192 + nt * 1024 + fragOff);
        accM[nt] = __builtin_amdgcn_wmma_f32_16x16x32_bf16(
            false, a, false, b, (short)0, accM[nt], false, false);
      }
    }
    {
      v16bf a = mk_frag<false>(ar + kb, s, ss);
#pragma unroll
      for (int nt = 0; nt < 4; ++nt) {
        v16bf b = *(const v16bf*)(wbuf + 12288 + nt * 1024 + fragOff);
        accM[nt] = __builtin_amdgcn_wmma_f32_16x16x32_bf16(
            false, a, false, b, (short)0, accM[nt], false, false);
      }
    }
    __syncthreads();       // everyone done with buf (kt&1) before re-filling
  }

  // ---- LayerNorm stats: combine the two half-rows (lane l and l+16) ----
  s  += __shfl_xor(s, 16, 32);
  ss += __shfl_xor(ss, 16, 32);
  const float mu  = s * (1.f / Hdim);
  const float inv = rsqrtf(ss * (1.f / Hdim) - mu * mu + LN_EPS);

  float cgv[4], cbv[4], btv[4];
#pragma unroll
  for (int nt = 0; nt < 4; ++nt) {
    cgv[nt] = cg[nt * 16 + n0];
    cbv[nt] = cb[nt * 16 + n0];
    btv[nt] = btot[nt * 16 + n0];
  }

  // ---- epilogue: LN correction -> logits -> probs in registers ----
  // C/D layout: VGPR r holds row r (lanes 0-15) or row 8+r (lanes 16-31).
  float P[4][8];
#pragma unroll
  for (int r = 0; r < 8; ++r) {
    int srcLane = (laneHi ? 8 : 0) + r;           // lane holding row stats
    float mu_r  = __shfl(mu,  srcLane, 16);
    float inv_r = __shfl(inv, srcLane, 16);
#pragma unroll
    for (int nt = 0; nt < 4; ++nt)
      P[nt][r] = inv_r * (accX[nt][r] - mu_r * cgv[nt]) + cbv[nt]
               + accM[nt][r] + btv[nt];
  }

  float epart[4] = {0.f, 0.f, 0.f, 0.f};

#pragma unroll
  for (int r = 0; r < 8; ++r) {
    // softmax over 64 experts: 4 local + 16-lane butterfly (stays in half-wave)
    float m4 = fmaxf(fmaxf(P[0][r], P[1][r]), fmaxf(P[2][r], P[3][r]));
#pragma unroll
    for (int o = 8; o >= 1; o >>= 1) m4 = fmaxf(m4, __shfl_xor(m4, o, 32));
    float t0 = __expf(P[0][r] - m4);
    float t1 = __expf(P[1][r] - m4);
    float t2 = __expf(P[2][r] - m4);
    float t3 = __expf(P[3][r] - m4);
    float sum4 = t0 + t1 + t2 + t3;
#pragma unroll
    for (int o = 8; o >= 1; o >>= 1) sum4 += __shfl_xor(sum4, o, 32);
    float rinv = 1.f / sum4;
    float q0 = t0 * rinv, q1 = t1 * rinv, q2 = t2 * rinv, q3 = t3 * rinv;
    epart[0] += q0; epart[1] += q1; epart[2] += q2; epart[3] += q3;

    // top-4: iterative arg-max, low-index tie-break (matches jax top_k)
    float tv[4]; int ti_[4]; float tsum = 0.f;
#pragma unroll
    for (int j = 0; j < 4; ++j) {
      float bv = q0; int bx = n0;
      if (q1 > bv) { bv = q1; bx = n0 + 16; }
      if (q2 > bv) { bv = q2; bx = n0 + 32; }
      if (q3 > bv) { bv = q3; bx = n0 + 48; }
#pragma unroll
      for (int o = 8; o >= 1; o >>= 1) {
        float ov = __shfl_xor(bv, o, 32);
        int   ox = __shfl_xor(bx, o, 32);
        if (ov > bv || (ov == bv && ox < bx)) { bv = ov; bx = ox; }
      }
      tv[j] = bv; ti_[j] = bx; tsum += bv;
      if      (bx == n0)      q0 = -1.f;
      else if (bx == n0 + 16) q1 = -1.f;
      else if (bx == n0 + 32) q2 = -1.f;
      else if (bx == n0 + 48) q3 = -1.f;
    }
    if (n0 == 0) {
      int grow = rowBase + (laneHi ? 8 : 0) + r;
      float rn = 1.f / tsum;
#pragma unroll
      for (int j = 0; j < 4; ++j) {
        outI[grow * 4 + j] = ti_[j];          // int32 expert indices
        outP[grow * 4 + j] = tv[j] * rn;      // renormalized top probs
      }
    }
  }

  // ---- per-expert prob sums (for aux loss): fold halves, one atomic each ----
#pragma unroll
  for (int nt = 0; nt < 4; ++nt) {
    epart[nt] += __shfl_xor(epart[nt], 16, 32);
    if (!laneHi) atomicAdd(&esum[nt * 16 + n0], epart[nt]);
  }
}

// ---------- aux loss scalar --------------------------------------------------
__global__ void aux_loss_kernel(const float* __restrict__ esum,
                                float* __restrict__ out) {
  if (threadIdx.x == 0) {
    float a = 0.f;
    for (int e = 0; e < Edim; ++e) {
      float p = esum[e] * (1.f / (float)Mrows);
      a += p * __logf(p * (float)Edim + 1e-9f);
    }
    out[0] = a;
  }
}

// ---------------------------------------------------------------------------
extern "C" void kernel_launch(void* const* d_in, const int* in_sizes, int n_in,
                              void* d_out, int out_size, void* d_ws, size_t ws_size,
                              hipStream_t stream) {
  (void)in_sizes; (void)n_in; (void)out_size; (void)ws_size;
  const float* x  = (const float*)d_in[0];
  const float* mi = (const float*)d_in[1];
  const float* mt = (const float*)d_in[2];
  const float* ma = (const float*)d_in[3];
  const float* ln_g = (const float*)d_in[4];
  const float* ln_b = (const float*)d_in[5];
  const float* Wg = (const float*)d_in[6];
  const float* bg = (const float*)d_in[7];
  const float* Wi = (const float*)d_in[8];
  const float* bi = (const float*)d_in[9];
  const float* Wt = (const float*)d_in[10];
  const float* bt = (const float*)d_in[11];
  const float* Wa = (const float*)d_in[12];
  const float* ba = (const float*)d_in[13];

  uint32_t* Wp   = (uint32_t*)d_ws + WS_WP;
  float*    cg   = (float*)d_ws + WS_CG;
  float*    cb   = (float*)d_ws + WS_CB;
  float*    btot = (float*)d_ws + WS_BTOT;
  float*    es   = (float*)d_ws + WS_ESUM;

  int*   outI = (int*)d_out;                     // top_i: 16384*4 int32
  float* outP = (float*)d_out + Mrows * 4;       // top_p: 16384*4 f32
  float* outA = (float*)d_out + Mrows * 8;       // aux loss scalar

  pack_weights_kernel<<<1024, 256, 0, stream>>>(Wg, Wi, Wt, Wa, ln_g, Wp);
  router_consts_kernel<<<1, 64, 0, stream>>>(Wg, ln_g, ln_b, bg, bi, bt, ba,
                                             cg, cb, btot, es);
  router_main_kernel<<<Mrows / 64, 128, 0, stream>>>(x, mi, mt, ma, Wp,
                                                     cg, cb, btot, es,
                                                     outI, outP);
  aux_loss_kernel<<<1, 32, 0, stream>>>(es, outA);
}
